// ResidualBlock_87591563034889
// MI455X (gfx1250) — compile-verified, matrix operand order checked by reading, not run on hardware
//
#include <hip/hip_runtime.h>
#include <hip/hip_bf16.h>
#include <math.h>

typedef __attribute__((ext_vector_type(16))) _Float16 v16h;
typedef __attribute__((ext_vector_type(8)))  float    v8f;

#define BB 64
#define NN 128
#define DD 128
#define BN (BB*NN)   // 8192
#define NG 50
#define KK 128       // every GEMM in this problem has K = 128
#define KT (KK/32)   // 4 k-tiles

static __device__ __forceinline__ float siluf(float x) { return x / (1.0f + __expf(-x)); }

// ---------------- WMMA fragment loaders (CDNA5 16x16x32 f16 layouts) ----------------
// A 16x32 (MxK) row-major: lane = half*16 + m ; VGPR v holds K = (v&3)*2 + (v>>2)*16 + half*8
// -> per lane two contiguous 8-half runs: [k0+half*8 .. +7] and [k0+half*8+16 .. +23]
__device__ __forceinline__ v16h load_frag_a(const _Float16* __restrict__ A, int lda,
                                            int row0, int k0, int lane) {
  int half = lane >> 4, m = lane & 15;
  const _Float16* rp = A + (size_t)(row0 + m) * lda + k0 + half * 8;
  v16h a;
#pragma unroll
  for (int v = 0; v < 8; ++v) {
    int kb = (v & 3) * 2 + (v >> 2) * 16;
    a[2 * v]     = rp[kb];
    a[2 * v + 1] = rp[kb + 1];
  }
  return a;
}

// B fragments are pre-packed: P[((kt*ntiles + ntg)*32 + lane)*16 + e] -> one 32B contiguous load
__device__ __forceinline__ v16h load_frag_b_packed(const _Float16* __restrict__ P,
                                                   int ntiles, int kt, int ntg, int lane) {
  return *(const v16h*)(P + ((((size_t)kt * ntiles + ntg) * 32 + lane) << 4));
}

// ---------------- generic / batched WMMA GEMM (K = 128 fixed) ----------------
// C(M x Nn) = act(A(M x 128) @ B(128 x Nn) + bias) [+ resid]
// grid: (Nn/128, M/128, batches); block 256 = 8 waves; each wave -> one 16 x 128 strip.
// act: 0 = none, 1 = relu, 2 = finite-clean (NaN/Inf -> 0, before resid add)
__global__ __launch_bounds__(256)
void wmma_gemm_kernel(const _Float16* __restrict__ A, const _Float16* __restrict__ Bp,
                      const float* __restrict__ bias, const float* __restrict__ resid,
                      float* __restrict__ Cf, _Float16* __restrict__ Ch,
                      int Nn,
                      long long sA, long long sB, long long sC, int act) {
  int b = blockIdx.z;
  const _Float16* Ab = A + (size_t)b * sA;
  const _Float16* Bb = Bp + (size_t)b * sB;
  int ntiles = Nn >> 4;
  int mBlock = blockIdx.y * 128;
  int nBlock = blockIdx.x * 128;
  int wave = threadIdx.x >> 5, lane = threadIdx.x & 31;
  int mt = mBlock + wave * 16;
  int nt0 = nBlock >> 4;

  v8f acc[8] = {};
#pragma unroll
  for (int kt = 0; kt < KT; ++kt) {
    v16h a = load_frag_a(Ab, KK, mt, kt * 32, lane);
#pragma unroll
    for (int nt = 0; nt < 8; ++nt) {
      v16h bf = load_frag_b_packed(Bb, ntiles, kt, nt0 + nt, lane);
      acc[nt] = __builtin_amdgcn_wmma_f32_16x16x32_f16(
          false, a, false, bf, (short)0, acc[nt], false, false);
    }
  }

  // C/D layout: VGPR r -> M = r + 8*half, N = lane&15
  int n0 = lane & 15, half = lane >> 4;
#pragma unroll
  for (int nt = 0; nt < 8; ++nt) {
    int col = nBlock + nt * 16 + n0;
    float bval = bias ? bias[col] : 0.0f;
#pragma unroll
    for (int r = 0; r < 8; ++r) {
      int row = mt + r + 8 * half;
      float v = acc[nt][r] + bval;
      if (act == 1) v = fmaxf(v, 0.0f);
      if (act == 2) v = __builtin_isfinite(v) ? v : 0.0f;
      size_t idx = (size_t)b * sC + (size_t)row * Nn + col;
      if (resid) v += resid[idx];
      if (Cf) Cf[idx] = v;
      if (Ch) Ch[idx] = (_Float16)v;
    }
  }
}

// ---------------- B-operand packers (into WMMA fragment order) ----------------
// One thread per (kt, nt, lane, v): writes 2 consecutive f16 of the fragment.
__global__ __launch_bounds__(256)
void pack_b_f32_kernel(const float* __restrict__ W, _Float16* __restrict__ P,
                       int K, int N) {
  int tid = blockIdx.x * 256 + threadIdx.x;
  if (tid >= K * N / 2) return;
  int v    = tid & 7;
  int lane = (tid >> 3) & 31;
  int tile = tid >> 8;
  int ntiles = N >> 4;
  int nt = tile % ntiles, kt = tile / ntiles;
  int k   = kt * 32 + (v & 3) * 2 + (v >> 2) * 16 + (lane >> 4) * 8;
  int col = nt * 16 + (lane & 15);
  size_t po = (((size_t)tile * 32 + lane) << 4) + 2 * v;
  P[po]     = (_Float16)W[(size_t)k * N + col];
  P[po + 1] = (_Float16)W[(size_t)(k + 1) * N + col];
}

__global__ __launch_bounds__(256)
void pack_b_f16_kernel(const _Float16* __restrict__ W, _Float16* __restrict__ P,
                       int K, int N, long long bstride) {
  int b = blockIdx.z;
  const _Float16* Wb = W + (size_t)b * bstride;
  _Float16* Pb = P + (size_t)b * bstride;
  int tid = blockIdx.x * 256 + threadIdx.x;
  if (tid >= K * N / 2) return;
  int v    = tid & 7;
  int lane = (tid >> 3) & 31;
  int tile = tid >> 8;
  int ntiles = N >> 4;
  int nt = tile % ntiles, kt = tile / ntiles;
  int k   = kt * 32 + (v & 3) * 2 + (v >> 2) * 16 + (lane >> 4) * 8;
  int col = nt * 16 + (lane & 15);
  size_t po = (((size_t)tile * 32 + lane) << 4) + 2 * v;
  Pb[po]     = Wb[(size_t)k * N + col];
  Pb[po + 1] = Wb[(size_t)(k + 1) * N + col];
}

// ---------------- RMSNorm: one wave per 128-wide row ----------------
__global__ __launch_bounds__(256)
void rmsnorm_kernel(const float* __restrict__ x, const float* __restrict__ w,
                    float* __restrict__ x1, _Float16* __restrict__ x1h) {
  int row = blockIdx.x * 8 + (threadIdx.x >> 5);
  int lane = threadIdx.x & 31;
  const float* xr = x + (size_t)row * DD;
  float v[4];
  float ss = 0.0f;
#pragma unroll
  for (int i = 0; i < 4; ++i) { v[i] = xr[lane + 32 * i]; ss += v[i] * v[i]; }
#pragma unroll
  for (int off = 16; off > 0; off >>= 1) ss += __shfl_xor(ss, off, 32);
  float r = rsqrtf(ss * (1.0f / DD) + 1e-5f);
#pragma unroll
  for (int i = 0; i < 4; ++i) {
    int c = lane + 32 * i;
    float o = v[i] * r * w[c];
    x1[(size_t)row * DD + c]  = o;
    x1h[(size_t)row * DD + c] = (_Float16)o;
  }
}

// ---------------- distances, edge mask, Gaussian smear -> em (A-role) and adj (f16) ----------------
__global__ __launch_bounds__(128)
void adj_kernel(const float* __restrict__ pos, const float* __restrict__ lin_w,
                const float* __restrict__ lin_b,
                _Float16* __restrict__ em_h, _Float16* __restrict__ adj_h) {
  int b = blockIdx.x, i = threadIdx.x;
  __shared__ float px[NN], py[NN], pz[NN], sw[NG];
  const float* p = pos + (size_t)b * NN * 3;
  px[i] = p[i * 3]; py[i] = p[i * 3 + 1]; pz[i] = p[i * 3 + 2];
  if (i < NG) sw[i] = lin_w[i];
  __syncthreads();
  float xi = px[i], yi = py[i], zi = pz[i];

  float deg = 0.0f;
  for (int j = 0; j < NN; ++j) {
    if (j == i) continue;
    float dx = xi - px[j], dy = yi - py[j], dz = zi - pz[j];
    float d2 = dx * dx + dy * dy + dz * dz;
    if (d2 < 100.0f) deg += 1.0f;
  }
  float invdeg = 1.0f / fmaxf(deg, 1.0f);

  float lb = lin_b[0];
  const float step  = 10.0f / 49.0f;
  const float coeff = -0.5f / (step * step);
  size_t base = ((size_t)b * NN + i) * NN;
  for (int j = 0; j < NN; ++j) {
    float dx = xi - px[j], dy = yi - py[j], dz = zi - pz[j];
    float d2 = dx * dx + dy * dy + dz * dz;
    bool edge = (j != i) && (d2 < 100.0f);
    if (edge) {
      float dist = sqrtf(d2);
      float dd = lb;
      for (int g = 0; g < NG; ++g) {
        float t = dist - (float)g * step;
        dd += sw[g] * __expf(coeff * t * t);
      }
      adj_h[base + j] = (_Float16)dd;
      em_h[base + j]  = (_Float16)invdeg;
    } else {
      adj_h[base + j] = (_Float16)0.0f;
      em_h[base + j]  = (_Float16)0.0f;
    }
  }
}

// ---------------- causal depthwise conv (D_CONV=4) + silu; xm = xr[:, :128] ----------------
__global__ __launch_bounds__(256)
void conv_silu_kernel(const float* __restrict__ xr, const float* __restrict__ conv_w,
                      const float* __restrict__ conv_b, float* __restrict__ xc) {
  int idx = blockIdx.x * 256 + threadIdx.x;  // BN*DD elements
  int d = idx & (DD - 1);
  int row = idx >> 7;
  int t = row & (NN - 1);
  float acc = conv_b[d];
#pragma unroll
  for (int j = 0; j < 4; ++j) {
    int tt = t + j - 3;
    if (tt >= 0) acc += xr[(size_t)(row + j - 3) * (2 * DD) + d] * conv_w[d * 4 + j];
  }
  xc[idx] = siluf(acc);
}

// ---------------- x_proj (N=12) + dt_proj (K=8) + softplus: one wave per row ----------------
__global__ __launch_bounds__(256)
void xproj_kernel(const float* __restrict__ xc, const float* __restrict__ x_proj_w,
                  const float* __restrict__ dt_proj_w, const float* __restrict__ dt_proj_b,
                  _Float16* __restrict__ delta_h, float* __restrict__ BC) {
  int row = blockIdx.x * 8 + (threadIdx.x >> 5);
  int lane = threadIdx.x & 31;
  const float* xrow = xc + (size_t)row * DD;
  float part[12];
#pragma unroll
  for (int c = 0; c < 12; ++c) part[c] = 0.0f;
#pragma unroll
  for (int q = 0; q < 4; ++q) {
    int k = lane * 4 + q;
    float xv = xrow[k];
#pragma unroll
    for (int c = 0; c < 12; ++c) part[c] += xv * x_proj_w[k * 12 + c];
  }
#pragma unroll
  for (int c = 0; c < 12; ++c)
    for (int off = 16; off > 0; off >>= 1) part[c] += __shfl_xor(part[c], off, 32);
#pragma unroll
  for (int q = 0; q < 4; ++q) {
    int d = lane + 32 * q;
    float dt = dt_proj_b[d];
#pragma unroll
    for (int r = 0; r < 8; ++r) dt += part[r] * dt_proj_w[r * DD + d];
    float sp = (dt > 20.0f) ? dt : log1pf(__expf(dt));
    delta_h[(size_t)row * DD + d] = (_Float16)sp;
  }
  if (lane == 0) {
    BC[(size_t)row * 4 + 0] = part[8];
    BC[(size_t)row * 4 + 1] = part[9];
    BC[(size_t)row * 4 + 2] = part[10];
    BC[(size_t)row * 4 + 3] = part[11];
  }
}

// ---------------- sequential SSM scan: 1 thread per channel, 1 block per batch ----------------
__global__ __launch_bounds__(128)
void scan_kernel(const float* __restrict__ dp, const float* __restrict__ xc,
                 const float* __restrict__ xr, const float* __restrict__ BC,
                 const float* __restrict__ A_log, const float* __restrict__ D_param,
                 _Float16* __restrict__ y_h) {
  int b = blockIdx.x, d = threadIdx.x;
  float A0 = -__expf(A_log[d * 2 + 0]);
  float A1 = -__expf(A_log[d * 2 + 1]);
  float Dp = D_param[d];
  float h0 = 0.0f, h1 = 0.0f;
  for (int t = 0; t < NN; ++t) {
    size_t row = (size_t)b * NN + t;
    float dpv = dp[row * DD + d];
    float xcv = xc[row * DD + d];
    float B0 = BC[row * 4 + 0], B1 = BC[row * 4 + 1];
    float C0 = BC[row * 4 + 2], C1 = BC[row * 4 + 3];
    h0 = __expf(dpv * A0) * h0 + dpv * B0 * xcv;
    h1 = __expf(dpv * A1) * h1 + dpv * B1 * xcv;
    float y = h0 * C0 + h1 * C1 + xcv * Dp;
    float r = xr[row * (2 * DD) + DD + d];  // res half
    y *= siluf(r);
    y_h[row * DD + d] = (_Float16)y;
  }
}

// ---------------- tail: pass-through outputs (pos, map_idx) ----------------
__global__ void tail_kernel(const float* __restrict__ pos, const int* __restrict__ map_idx,
                            float* __restrict__ out) {
  int i = blockIdx.x * 256 + threadIdx.x;
  if (i < BN * 3) out[(size_t)BN * DD + i] = pos[i];
  if (i < BN)     out[(size_t)BN * DD + BN * 3 + i] = (float)map_idx[i];
}

extern "C" void kernel_launch(void* const* d_in, const int* in_sizes, int n_in,
                              void* d_out, int out_size, void* d_ws, size_t ws_size,
                              hipStream_t stream) {
  (void)in_sizes; (void)n_in; (void)out_size; (void)ws_size;
  const float* x          = (const float*)d_in[0];
  const float* pos        = (const float*)d_in[1];
  const int*   map_idx    = (const int*)d_in[2];
  /* d_in[3] = batch (unused) */
  const float* norm_w     = (const float*)d_in[4];
  const float* gnn_w1     = (const float*)d_in[5];
  const float* gnn_b1     = (const float*)d_in[6];
  const float* gnn_w2     = (const float*)d_in[7];
  const float* gnn_b2     = (const float*)d_in[8];
  const float* lin_w      = (const float*)d_in[9];
  const float* lin_b      = (const float*)d_in[10];
  const float* in_proj_w  = (const float*)d_in[11];
  const float* conv_w     = (const float*)d_in[12];
  const float* conv_b     = (const float*)d_in[13];
  const float* x_proj_w   = (const float*)d_in[14];
  const float* dt_proj_w  = (const float*)d_in[15];
  const float* dt_proj_b  = (const float*)d_in[16];
  const float* A_log      = (const float*)d_in[17];
  const float* D_param    = (const float*)d_in[18];
  const float* out_proj_w = (const float*)d_in[19];
  float* out = (float*)d_out;

  // Workspace carve-up (~42 MB; fully resident in MI455X's 192 MB L2)
  char* ws = (char*)d_ws;
  size_t off = 0;
  auto alloc = [&](size_t bytes) -> char* {
    char* p = ws + off;
    off += (bytes + 255) & ~(size_t)255;
    return p;
  };
  float*    x1    = (float*)alloc((size_t)BN * DD * 4);
  _Float16* x1h   = (_Float16*)alloc((size_t)BN * DD * 2);
  _Float16* w1p   = (_Float16*)alloc((size_t)DD * DD * 2);
  _Float16* w2p   = (_Float16*)alloc((size_t)DD * DD * 2);
  _Float16* ipwp  = (_Float16*)alloc((size_t)DD * 2 * DD * 2);
  _Float16* opwp  = (_Float16*)alloc((size_t)DD * DD * 2);
  _Float16* em_h  = (_Float16*)alloc((size_t)BB * NN * NN * 2);
  _Float16* adj_h = (_Float16*)alloc((size_t)BB * NN * NN * 2);
  _Float16* adj_p = (_Float16*)alloc((size_t)BB * NN * NN * 2);
  _Float16* h_h   = (_Float16*)alloc((size_t)BN * DD * 2);
  _Float16* h_p   = (_Float16*)alloc((size_t)BN * DD * 2);
  _Float16* agg_h = (_Float16*)alloc((size_t)BN * DD * 2);
  _Float16* x2h   = (_Float16*)alloc((size_t)BN * DD * 2);
  float*    xrws  = (float*)alloc((size_t)BN * 2 * DD * 4);
  float*    xc    = (float*)alloc((size_t)BN * DD * 4);
  _Float16* dlt_h = (_Float16*)alloc((size_t)BN * DD * 2);
  float*    BC    = (float*)alloc((size_t)BN * 4 * 4);
  float*    dp    = (float*)alloc((size_t)BN * DD * 4);
  _Float16* y_h   = (_Float16*)alloc((size_t)BN * DD * 2);

  // weights -> packed f16 B-fragments
  pack_b_f32_kernel<<<dim3(32), 256, 0, stream>>>(gnn_w1, w1p, DD, DD);
  pack_b_f32_kernel<<<dim3(32), 256, 0, stream>>>(gnn_w2, w2p, DD, DD);
  pack_b_f32_kernel<<<dim3(64), 256, 0, stream>>>(in_proj_w, ipwp, DD, 2 * DD);
  pack_b_f32_kernel<<<dim3(32), 256, 0, stream>>>(out_proj_w, opwp, DD, DD);

  rmsnorm_kernel<<<dim3(BN / 8), 256, 0, stream>>>(x, norm_w, x1, x1h);
  adj_kernel<<<dim3(BB), 128, 0, stream>>>(pos, lin_w, lin_b, em_h, adj_h);
  pack_b_f16_kernel<<<dim3(32, 1, BB), 256, 0, stream>>>(adj_h, adj_p, NN, NN,
                                                         (long long)NN * NN);

  // h = relu(x1 @ gnn_w1 + b1)
  wmma_gemm_kernel<<<dim3(1, BN / 128, 1), 256, 0, stream>>>(
      x1h, w1p, gnn_b1, nullptr, nullptr, h_h, DD, 0, 0, 0, 1);
  pack_b_f16_kernel<<<dim3(32, 1, BB), 256, 0, stream>>>(h_h, h_p, NN, DD,
                                                         (long long)NN * DD);
  // agg = (em/deg) @ h   (batched: per-graph A and B)
  wmma_gemm_kernel<<<dim3(1, 1, BB), 256, 0, stream>>>(
      em_h, h_p, nullptr, nullptr, nullptr, agg_h, DD,
      (long long)NN * NN, (long long)NN * DD, (long long)NN * DD, 0);
  // x2 = agg @ gnn_w2 + b2
  wmma_gemm_kernel<<<dim3(1, BN / 128, 1), 256, 0, stream>>>(
      agg_h, w2p, gnn_b2, nullptr, nullptr, x2h, DD, 0, 0, 0, 0);
  // xr = x2 @ in_proj_w  (N = 256)
  wmma_gemm_kernel<<<dim3(2, BN / 128, 1), 256, 0, stream>>>(
      x2h, ipwp, nullptr, nullptr, xrws, nullptr, 2 * DD, 0, 0, 0, 0);

  conv_silu_kernel<<<dim3(BN * DD / 256), 256, 0, stream>>>(xrws, conv_w, conv_b, xc);
  xproj_kernel<<<dim3(BN / 8), 256, 0, stream>>>(xc, x_proj_w, dt_proj_w, dt_proj_b, dlt_h, BC);

  // delta_p = delta @ adj  (batched; adj_pad == adj since m == D)
  wmma_gemm_kernel<<<dim3(1, 1, BB), 256, 0, stream>>>(
      dlt_h, adj_p, nullptr, nullptr, dp, nullptr, DD,
      (long long)NN * DD, (long long)NN * NN, (long long)NN * DD, 0);

  scan_kernel<<<dim3(BB), 128, 0, stream>>>(dp, xc, xrws, BC, A_log, D_param, y_h);

  // out = finite_clean(y @ out_proj) + x1
  wmma_gemm_kernel<<<dim3(1, BN / 128, 1), 256, 0, stream>>>(
      y_h, opwp, nullptr, x1, out, nullptr, DD, 0, 0, 0, 2);

  tail_kernel<<<dim3((BN * 3 + 255) / 256), 256, 0, stream>>>(pos, map_idx, out);
}
